// CrossPath_V2_11682311045318
// MI455X (gfx1250) — matrix operand order repaired and checked by reading, not compiled, not run on hardware
//
#include <hip/hip_runtime.h>
#include <hip/hip_bf16.h>

// CrossPath_V2 for MI455X (gfx1250): bf16 WMMA pipeline.
// - fragment-major ("swizzled") LDS layout for B tiles: B frag = 2x ds_load_b128
// - software-pipelined K loops: ping-pong LDS + register prefetch of next tile,
//   one barrier per K-step (buffer reuse is 2 iterations away).
// - global_prefetch on streamed operands.
// B=4, N=4096, C=512, H=8, D=64.

typedef __bf16 bf16_t;
typedef __attribute__((ext_vector_type(16))) __bf16 v16bf;
typedef __attribute__((ext_vector_type(8)))  __bf16 v8bf;
typedef __attribute__((ext_vector_type(4)))  __bf16 v4bf;
typedef __attribute__((ext_vector_type(8)))  float  v8f;

__device__ __forceinline__ v8f wmma_bf16(v16bf a, v16bf b, v8f c) {
  return __builtin_amdgcn_wmma_f32_16x16x32_bf16(
      false, a, false, b, (short)0, c, false, false);
}

// A fragment: 16x32 (MxK), row-major LDS, row stride ld (multiple of 8).
__device__ __forceinline__ v16bf load_frag_a(const bf16_t* __restrict__ s, int ld) {
  const int lane = threadIdx.x & 31;
  const bf16_t* p = s + (lane & 15) * ld + ((lane >> 4) << 4);
  const v8bf lo = *(const v8bf*)p;
  const v8bf hi = *(const v8bf*)(p + 8);
  return __builtin_shufflevector(lo, hi, 0,1,2,3,4,5,6,7,8,9,10,11,12,13,14,15);
}

// Swizzled B tile (BK=32 x BN): element (k,n) stored so each lane's 16
// K-values are contiguous (32 bytes).
__device__ __forceinline__ int bswz(int k, int n) {
  return ((((n >> 4) << 1) + (k >> 4)) << 8) + ((n & 15) << 4) + (k & 15);
}
__device__ __forceinline__ v16bf load_frag_b_swz(const bf16_t* __restrict__ s, int ntile) {
  const int lane = threadIdx.x & 31;
  const bf16_t* p = s + ((((ntile << 1) + (lane >> 4)) << 8) + ((lane & 15) << 4));
  return *(const v16bf*)p;
}

// ---------------------------------------------------------------------------
// Generic GEMM: out[M,N](bf16) = epi(A[M,K] @ Bw[K,N](f32) + bias)
// Block tile 64x128, BK=32, 8 waves; wave -> 16 rows x 64 cols.
// Ping-pong LDS + register prefetch: one barrier per K-step.
// ---------------------------------------------------------------------------
template <bool A_F32, bool RELU_BIAS>
__global__ void __launch_bounds__(256)
gemm_bf16_out(const void* __restrict__ Aptr, const float* __restrict__ Bw,
              const float* __restrict__ bias, bf16_t* __restrict__ out,
              int M, int N, int K) {
  constexpr int BM = 64, BN = 128, BK = 32;
  constexpr int LDA = BK + 8;
  __shared__ __align__(16) bf16_t As[2][BM * LDA];
  __shared__ __align__(32) bf16_t Bs[2][BK * BN];

  const int tid = threadIdx.x;
  const int wid = tid >> 5;
  const int wr = wid >> 1;   // 0..3 : 16-row tile
  const int wc = wid & 1;    // 0..1 : 64-col half
  const int m0 = blockIdx.y * BM;
  const int n0 = blockIdx.x * BN;

  const float*  Af = (const float*)Aptr;
  const bf16_t* Ab = (const bf16_t*)Aptr;

  // fixed per-thread staging coordinates
  int ar[2], ac[2], br[4], bc[4];
#pragma unroll
  for (int u = 0; u < 2; ++u) {
    const int idx = (tid + u * 256) << 2;
    ar[u] = idx >> 5; ac[u] = idx & 31;
  }
#pragma unroll
  for (int u = 0; u < 4; ++u) {
    const int idx = (tid + u * 256) << 2;
    br[u] = idx >> 7; bc[u] = idx & 127;
  }

  float4 aF[2]; v4bf aB[2]; float4 bF[4];
  auto fetch = [&](int k0) {
#pragma unroll
    for (int u = 0; u < 2; ++u) {
      if constexpr (A_F32)
        aF[u] = *(const float4*)&Af[(size_t)(m0 + ar[u]) * K + k0 + ac[u]];
      else
        aB[u] = *(const v4bf*)&Ab[(size_t)(m0 + ar[u]) * K + k0 + ac[u]];
    }
#pragma unroll
    for (int u = 0; u < 4; ++u)
      bF[u] = *(const float4*)&Bw[(size_t)(k0 + br[u]) * N + n0 + bc[u]];
  };
  auto stage = [&](int p) {
#pragma unroll
    for (int u = 0; u < 2; ++u) {
      bf16_t* q = &As[p][ar[u] * LDA + ac[u]];
      if constexpr (A_F32) {
        q[0] = (bf16_t)aF[u].x; q[1] = (bf16_t)aF[u].y;
        q[2] = (bf16_t)aF[u].z; q[3] = (bf16_t)aF[u].w;
      } else {
        *(v4bf*)q = aB[u];
      }
    }
#pragma unroll
    for (int u = 0; u < 4; ++u) {
      Bs[p][bswz(br[u], bc[u])]     = (bf16_t)bF[u].x;
      Bs[p][bswz(br[u], bc[u] + 1)] = (bf16_t)bF[u].y;
      Bs[p][bswz(br[u], bc[u] + 2)] = (bf16_t)bF[u].z;
      Bs[p][bswz(br[u], bc[u] + 3)] = (bf16_t)bF[u].w;
    }
  };

  v8f acc[4] = {};
  fetch(0);
  int p = 0;
  for (int k0 = 0; k0 < K; k0 += BK) {
    stage(p);
    __syncthreads();
    if (k0 + BK < K) {
      fetch(k0 + BK);                              // overlaps the WMMAs below
      if (k0 + 2 * BK < K)                         // prefetch 2 tiles ahead
        __builtin_prefetch(&Bw[(size_t)(k0 + 2 * BK) * N + n0 + (tid & 31) * 4], 0, 1);
    }
    const v16bf a = load_frag_a(As[p] + wr * 16 * LDA, LDA);
#pragma unroll
    for (int t = 0; t < 4; ++t)
      acc[t] = wmma_bf16(a, load_frag_b_swz(Bs[p], (wc << 2) + t), acc[t]);
    p ^= 1;
  }

  const int lane = tid & 31;
  const int rb = m0 + wr * 16 + ((lane >> 4) << 3);
#pragma unroll
  for (int t = 0; t < 4; ++t) {
    const int col = n0 + (wc << 6) + (t << 4) + (lane & 15);
#pragma unroll
    for (int j = 0; j < 8; ++j) {
      float v = acc[t][j];
      if constexpr (RELU_BIAS) {
        v += bias[col];
        v = v > 0.f ? v : 0.f;
      }
      out[(size_t)(rb + j) * N + col] = (bf16_t)v;
    }
  }
}

// ---------------------------------------------------------------------------
// ctx[bh] = softmax_over_rows( 0.125 * k^T @ v ), 64x64 per (b,h).
// K=4096 in 32-chunks, ping-pong LDS + register prefetch.
// ---------------------------------------------------------------------------
__global__ void __launch_bounds__(256)
ctx_kernel(const bf16_t* __restrict__ kv, bf16_t* __restrict__ ctx) {
  constexpr int LDA = 32 + 8, LDC = 65;
  __shared__ __align__(16) bf16_t As[2][64 * LDA];  // [d][nn] (k^T)
  __shared__ __align__(32) bf16_t Bs[2][32 * 64];   // swizzled v chunk
  __shared__ float Cs[64 * LDC];

  const int bh = blockIdx.x;
  const int b = bh >> 3, h = bh & 7;
  const int tid = threadIdx.x;
  const int wid = tid >> 5;
  const int wr = wid >> 1, wc = wid & 1;
  const size_t rowbase = (size_t)b * 4096;

  int kd[2], kn[2], ve[2], vn[2];
#pragma unroll
  for (int u = 0; u < 2; ++u) {
    const int idx = (tid + u * 256) << 2;
    kd[u] = idx & 63; kn[u] = idx >> 6;   // k^T staging coords
    ve[u] = idx & 63; vn[u] = idx >> 6;   // v staging coords
  }

  v4bf ka[2], va[2];
  auto fetch = [&](int n0) {
#pragma unroll
    for (int u = 0; u < 2; ++u)
      ka[u] = *(const v4bf*)&kv[((rowbase + n0 + kn[u]) << 10) + h * 64 + kd[u]];
#pragma unroll
    for (int u = 0; u < 2; ++u)
      va[u] = *(const v4bf*)&kv[((rowbase + n0 + vn[u]) << 10) + 512 + h * 64 + ve[u]];
  };
  auto stage = [&](int p) {
#pragma unroll
    for (int u = 0; u < 2; ++u) {
      As[p][(kd[u] + 0) * LDA + kn[u]] = ka[u][0];
      As[p][(kd[u] + 1) * LDA + kn[u]] = ka[u][1];
      As[p][(kd[u] + 2) * LDA + kn[u]] = ka[u][2];
      As[p][(kd[u] + 3) * LDA + kn[u]] = ka[u][3];
    }
#pragma unroll
    for (int u = 0; u < 2; ++u) {
      Bs[p][bswz(vn[u], ve[u])]     = va[u][0];
      Bs[p][bswz(vn[u], ve[u] + 1)] = va[u][1];
      Bs[p][bswz(vn[u], ve[u] + 2)] = va[u][2];
      Bs[p][bswz(vn[u], ve[u] + 3)] = va[u][3];
    }
  };

  v8f acc0 = {}, acc1 = {};
  fetch(0);
  int p = 0;
  for (int n0 = 0; n0 < 4096; n0 += 32) {
    stage(p);
    __syncthreads();
    if (n0 + 32 < 4096) {
      fetch(n0 + 32);
      if (n0 + 64 < 4096)
        __builtin_prefetch(&kv[((rowbase + n0 + 64 + (tid >> 3)) << 10) + h * 64], 0, 1);
    }
    const v16bf a = load_frag_a(As[p] + wr * 16 * LDA, LDA);
    acc0 = wmma_bf16(a, load_frag_b_swz(Bs[p], (wc << 1)), acc0);
    acc1 = wmma_bf16(a, load_frag_b_swz(Bs[p], (wc << 1) + 1), acc1);
    p ^= 1;
  }

  const int lane = tid & 31;
  const int cb = (wc << 5) + (lane & 15);
  const int rb = wr * 16 + ((lane >> 4) << 3);
#pragma unroll
  for (int j = 0; j < 8; ++j) {
    Cs[(rb + j) * LDC + cb]      = acc0[j] * 0.125f;  // scale = D^-0.5
    Cs[(rb + j) * LDC + cb + 16] = acc1[j] * 0.125f;
  }
  __syncthreads();

  // softmax over rows (d) per column e
  if (tid < 64) {
    const int e = tid;
    float mx = -3.4e38f;
    for (int d = 0; d < 64; ++d) mx = fmaxf(mx, Cs[d * LDC + e]);
    float sum = 0.f;
    for (int d = 0; d < 64; ++d) {
      const float ex = __expf(Cs[d * LDC + e] - mx);
      Cs[d * LDC + e] = ex;
      sum += ex;
    }
    const float inv = 1.f / sum;
    for (int d = 0; d < 64; ++d)
      ctx[((size_t)bh << 12) + d * 64 + e] = (bf16_t)(Cs[d * LDC + e] * inv);
  }
}

// ---------------------------------------------------------------------------
// a[b,n,h*64+e] = sum_d q[b,n,h*64+d] * ctx[bh][d][e]. K=64.
// 128 rows x 64 cols per (nblock, bh); 8 waves x (16 rows x 64 cols).
// ---------------------------------------------------------------------------
__global__ void __launch_bounds__(256)
attn_kernel(const bf16_t* __restrict__ cr, const bf16_t* __restrict__ ctx,
            bf16_t* __restrict__ abuf) {
  constexpr int LDQ = 64 + 8;
  __shared__ __align__(16) bf16_t Qs[128 * LDQ];
  __shared__ __align__(32) bf16_t Cs[64 * 64];   // 2 swizzled 32x64 k-blocks

  const int bh = blockIdx.y;
  const int b = bh >> 3, h = bh & 7;
  const int n0 = blockIdx.x * 128;
  const int tid = threadIdx.x;
  const int wid = tid >> 5;
  const size_t rowbase = (size_t)b * 4096 + n0;

  for (int i = tid; i < (128 * 64) / 4; i += 256) {
    const int idx = i << 2;
    const int d = idx & 63, r = idx >> 6;
    *(v4bf*)&Qs[r * LDQ + d] = *(const v4bf*)&cr[((rowbase + r) << 9) + h * 64 + d];
  }
  for (int i = tid; i < (64 * 64) / 4; i += 256) {
    const int idx = i << 2;
    const int e = idx & 63, d = idx >> 6;
    const v4bf t = *(const v4bf*)&ctx[((size_t)bh << 12) + idx];
    bf16_t* base = &Cs[(d >> 5) << 11];
    base[bswz(d & 31, e)]     = t[0];
    base[bswz(d & 31, e + 1)] = t[1];
    base[bswz(d & 31, e + 2)] = t[2];
    base[bswz(d & 31, e + 3)] = t[3];
  }
  __syncthreads();

  v8f acc[4] = {};
#pragma unroll
  for (int kk = 0; kk < 64; kk += 32) {
    const v16bf a = load_frag_a(Qs + wid * 16 * LDQ + kk, LDQ);
#pragma unroll
    for (int t = 0; t < 4; ++t)
      acc[t] = wmma_bf16(a, load_frag_b_swz(Cs + ((kk >> 5) << 11), t), acc[t]);
  }

  const int lane = tid & 31;
  const int rb = wid * 16 + ((lane >> 4) << 3);
#pragma unroll
  for (int t = 0; t < 4; ++t) {
    const int e = (t << 4) + (lane & 15);
#pragma unroll
    for (int j = 0; j < 8; ++j)
      abuf[((rowbase + rb + j) << 9) + h * 64 + e] = (bf16_t)acc[t][j];
  }
}

// ---------------------------------------------------------------------------
// out = LayerNorm( x + [s | a] @ We + be ) * g + bt ; K=1024, N=512.
// WG owns 16 full rows; LN fused via __shfl_xor + LDS cross-wave reduction.
// ---------------------------------------------------------------------------
__global__ void __launch_bounds__(256)
endproj_ln_kernel(const float* __restrict__ x, const float* __restrict__ s,
                  const bf16_t* __restrict__ a, const float* __restrict__ We,
                  const float* __restrict__ be, const float* __restrict__ g,
                  const float* __restrict__ bt, float* __restrict__ out) {
  constexpr int BK = 32;
  constexpr int LDA = BK + 8;
  __shared__ __align__(16) bf16_t As[16 * LDA];
  __shared__ __align__(32) bf16_t Bs[BK * 512];
  __shared__ float redS[8][16];
  __shared__ float redQ[8][16];
  __shared__ float meanArr[16], rstdArr[16];

  const int tid = threadIdx.x;
  const int wid = tid >> 5;
  const int lane = tid & 31;
  const int m0 = blockIdx.x * 16;

  v8f acc[4] = {};

  for (int k0 = 0; k0 < 1024; k0 += BK) {
    __syncthreads();
    if (k0 + BK < 1024)  // prefetch next We tile while this step runs
      __builtin_prefetch(&We[(size_t)(k0 + BK) * 512 + tid * 4], 0, 1);
    // A tile 16x32: y = [ s (f32, k<512) | a (bf16, k>=512) ]; branch uniform per k0
    for (int i = tid; i < (16 * BK) / 4; i += 256) {
      const int idx = i << 2;
      const int r = idx >> 5, c = idx & 31;
      const int k = k0 + c;
      if (k < 512) {
        const float4 v = *(const float4*)&s[(size_t)(m0 + r) * 512 + k];
        bf16_t* q = &As[r * LDA + c];
        q[0] = (bf16_t)v.x; q[1] = (bf16_t)v.y; q[2] = (bf16_t)v.z; q[3] = (bf16_t)v.w;
      } else {
        *(v4bf*)&As[r * LDA + c] =
            *(const v4bf*)&a[(size_t)(m0 + r) * 512 + (k - 512)];
      }
    }
    // B tile 32x512 from f32 We, swizzled
    for (int i = tid; i < (BK * 512) / 4; i += 256) {
      const int idx = i << 2;
      const int r = idx >> 9, c = idx & 511;
      const float4 w = *(const float4*)&We[(size_t)(k0 + r) * 512 + c];
      Bs[bswz(r, c)]     = (bf16_t)w.x;
      Bs[bswz(r, c + 1)] = (bf16_t)w.y;
      Bs[bswz(r, c + 2)] = (bf16_t)w.z;
      Bs[bswz(r, c + 3)] = (bf16_t)w.w;
    }
    __syncthreads();

    const v16bf av = load_frag_a(As, LDA);
#pragma unroll
    for (int t = 0; t < 4; ++t)
      acc[t] = wmma_bf16(av, load_frag_b_swz(Bs, (wid << 2) + t), acc[t]);
  }

  // epilogue: v += x + be ; per-row sum / sumsq
  const int rb = ((lane >> 4) << 3);
  float ps[8], pq[8];
#pragma unroll
  for (int j = 0; j < 8; ++j) { ps[j] = 0.f; pq[j] = 0.f; }
#pragma unroll
  for (int t = 0; t < 4; ++t) {
    const int col = (wid << 6) + (t << 4) + (lane & 15);
#pragma unroll
    for (int j = 0; j < 8; ++j) {
      float v = acc[t][j] + x[(size_t)(m0 + rb + j) * 512 + col] + be[col];
      acc[t][j] = v;
      ps[j] += v;
      pq[j] += v * v;
    }
  }
#pragma unroll
  for (int m = 1; m < 16; m <<= 1) {
#pragma unroll
    for (int j = 0; j < 8; ++j) {
      ps[j] += __shfl_xor(ps[j], m, 32);
      pq[j] += __shfl_xor(pq[j], m, 32);
    }
  }
  if ((lane & 15) == 0) {
#pragma unroll
    for (int j = 0; j < 8; ++j) {
      redS[wid][rb + j] = ps[j];
      redQ[wid][rb + j] = pq[j];
    }
  }
  __syncthreads();
  if (tid < 16) {
    float sS = 0.f, sQ = 0.f;
#pragma unroll
    for (int w = 0; w < 8; ++w) { sS += redS[w][tid]; sQ += redQ[w][tid]; }
    const float mean = sS * (1.f / 512.f);
    const float var  = sQ * (1.f / 512.f) - mean * mean;
    meanArr[tid] = mean;
    rstdArr[tid] = rsqrtf(var + 1e-5f);
  }
  __syncthreads();

#pragma unroll
  for (int t = 0; t < 4; ++t) {
    const int col = (wid << 6) + (t << 4) + (lane & 15);
#pragma unroll
    for (int j = 0; j < 8; ++j) {
      const int r = rb + j;
      out[(size_t)(m0 + r) * 512 + col] =
          (acc[t][j] - meanArr[r]) * rstdArr[r] * g[col] + bt[col];
    }
  }
}

// ---------------------------------------------------------------------------
extern "C" void kernel_launch(void* const* d_in, const int* in_sizes, int n_in,
                              void* d_out, int out_size, void* d_ws, size_t ws_size,
                              hipStream_t stream) {
  (void)in_sizes; (void)n_in; (void)out_size; (void)ws_size;

  const float* x1  = (const float*)d_in[0];
  const float* x2  = (const float*)d_in[1];
  const float* c1  = (const float*)d_in[2];
  const float* c2  = (const float*)d_in[3];
  const float* s1  = (const float*)d_in[4];
  const float* s2  = (const float*)d_in[5];
  const float* W3  = (const float*)d_in[6];
  const float* b3  = (const float*)d_in[7];
  const float* W4  = (const float*)d_in[8];
  const float* b4  = (const float*)d_in[9];
  const float* Wkv1 = (const float*)d_in[10];
  const float* Wkv2 = (const float*)d_in[11];
  const float* We1 = (const float*)d_in[12];
  const float* be1 = (const float*)d_in[13];
  const float* We2 = (const float*)d_in[14];
  const float* be2 = (const float*)d_in[15];
  const float* g1  = (const float*)d_in[16];
  const float* bt1 = (const float*)d_in[17];
  const float* g2  = (const float*)d_in[18];
  const float* bt2 = (const float*)d_in[19];

  const size_t M = 16384;  // B*N
  char* ws = (char*)d_ws;
  size_t off = 0;
  auto carve = [&](size_t bytes) -> void* {
    void* p = ws + off;
    off += (bytes + 255) & ~(size_t)255;
    return p;
  };
  bf16_t* c1r  = (bf16_t*)carve(M * 512 * 2);
  bf16_t* c2r  = (bf16_t*)carve(M * 512 * 2);
  bf16_t* kv1  = (bf16_t*)carve(M * 1024 * 2);
  bf16_t* kv2  = (bf16_t*)carve(M * 1024 * 2);
  bf16_t* ctx1 = (bf16_t*)carve(32 * 64 * 64 * 2);
  bf16_t* ctx2 = (bf16_t*)carve(32 * 64 * 64 * 2);
  bf16_t* a1b  = (bf16_t*)carve(M * 512 * 2);
  bf16_t* a2b  = (bf16_t*)carve(M * 512 * 2);

  const dim3 blk(256);

  // channel projections + ReLU (f32 in -> bf16 activations)
  gemm_bf16_out<true, true><<<dim3(4, 256), blk, 0, stream>>>(c1, W3, b3, c1r, 16384, 512, 512);
  gemm_bf16_out<true, true><<<dim3(4, 256), blk, 0, stream>>>(c2, W4, b4, c2r, 16384, 512, 512);

  // kv projections (bf16 in -> bf16 out)
  gemm_bf16_out<false, false><<<dim3(8, 256), blk, 0, stream>>>(c1r, Wkv1, nullptr, kv1, 16384, 1024, 512);
  gemm_bf16_out<false, false><<<dim3(8, 256), blk, 0, stream>>>(c2r, Wkv2, nullptr, kv2, 16384, 1024, 512);

  // ctx = softmax(scale * k^T v) per (b,h)
  ctx_kernel<<<32, blk, 0, stream>>>(kv1, ctx1);
  ctx_kernel<<<32, blk, 0, stream>>>(kv2, ctx2);

  // cross attention: a1 = q1 @ ctx2, a2 = q2 @ ctx1
  attn_kernel<<<dim3(32, 32), blk, 0, stream>>>(c1r, ctx2, a1b);
  attn_kernel<<<dim3(32, 32), blk, 0, stream>>>(c2r, ctx1, a2b);

  // end projection + residual + LayerNorm (fused)
  float* out1 = (float*)d_out;
  float* out2 = out1 + M * 512;
  endproj_ln_kernel<<<1024, blk, 0, stream>>>(x1, s1, a1b, We1, be1, g1, bt1, out1);
  endproj_ln_kernel<<<1024, blk, 0, stream>>>(x2, s2, a2b, We2, be2, g2, bt2, out2);
}
